// MultiHeadAttention_13348758356378
// MI455X (gfx1250) — matrix operand order repaired and checked by reading, not compiled, not run on hardware
//
#include <hip/hip_runtime.h>
#include <hip/hip_bf16.h>
#include <stdint.h>

// ---------------------------------------------------------------------------
// MHA forward for MI455X (gfx1250): bf16 WMMA 16x16x32, f32 accumulate.
// - contiguous 16B LDS fragment reads (ds_load_b128) into WMMA operand blocks
// - DPP16 butterfly reductions for softmax (no ds_bpermute)
// - double-buffered global->LDS staging via GLOBAL_LOAD_ASYNC_TO_LDS_B128
//   (ASYNCcnt), zero staging registers -> no scratch spills
// ---------------------------------------------------------------------------

#define D_MODEL   1024
#define NUM_HEADS 16
#define D_K       64
#define SEQ       2048
#define BATCH     2
#define ROWS      (BATCH * SEQ)   // 4096 token rows

typedef __attribute__((ext_vector_type(16))) __bf16 bf16x16;
typedef __attribute__((ext_vector_type(8)))  float  f32x8;
typedef __attribute__((ext_vector_type(4)))  int    v4i;

union FragBF { bf16x16 v; uint32_t u[8]; uint4 q[2]; };
union FragF  { f32x8   v; float    f[8]; };

#if __has_builtin(__builtin_amdgcn_global_load_async_to_lds_b128)
#define HAVE_ASYNC_LDS 1
#define AS1 __attribute__((address_space(1)))
#define AS3 __attribute__((address_space(3)))
__device__ __forceinline__ void async_copy16(const void* g, void* l) {
  __builtin_amdgcn_global_load_async_to_lds_b128((AS1 v4i*)g, (AS3 v4i*)l,
                                                 0, 0);
}
#else
#define HAVE_ASYNC_LDS 0
#endif

__device__ __forceinline__ void wait_async0() {
#if __has_builtin(__builtin_amdgcn_s_wait_asynccnt)
  __builtin_amdgcn_s_wait_asynccnt(0);
#else
  asm volatile("s_wait_asynccnt 0x0" ::: "memory");
#endif
}

__device__ __forceinline__ uint16_t f2bf(float f) {
  uint32_t u = __float_as_uint(f);
  u += 0x7fffu + ((u >> 16) & 1u);   // round to nearest even
  return (uint16_t)(u >> 16);
}

// 16-lane butterfly all-reduce via DPP16 (xor masks 1,3,7,15 span all 16
// lanes; mirrors/quad-perms stay inside each 16-lane row -> wave32 halves).
__device__ __forceinline__ float red16_max(float x) {
  float y;
  y = __int_as_float(__builtin_amdgcn_update_dpp(0, __float_as_int(x), 0x0B1, 0xf, 0xf, true));
  x = fmaxf(x, y);   // quad_perm [1,0,3,2]  : xor 1
  y = __int_as_float(__builtin_amdgcn_update_dpp(0, __float_as_int(x), 0x01B, 0xf, 0xf, true));
  x = fmaxf(x, y);   // quad_perm [3,2,1,0]  : xor 3
  y = __int_as_float(__builtin_amdgcn_update_dpp(0, __float_as_int(x), 0x141, 0xf, 0xf, true));
  x = fmaxf(x, y);   // row_half_mirror      : xor 7
  y = __int_as_float(__builtin_amdgcn_update_dpp(0, __float_as_int(x), 0x140, 0xf, 0xf, true));
  x = fmaxf(x, y);   // row_mirror           : xor 15
  return x;
}
__device__ __forceinline__ float red16_sum(float x) {
  float y;
  y = __int_as_float(__builtin_amdgcn_update_dpp(0, __float_as_int(x), 0x0B1, 0xf, 0xf, true));
  x += y;
  y = __int_as_float(__builtin_amdgcn_update_dpp(0, __float_as_int(x), 0x01B, 0xf, 0xf, true));
  x += y;
  y = __int_as_float(__builtin_amdgcn_update_dpp(0, __float_as_int(x), 0x141, 0xf, 0xf, true));
  x += y;
  y = __int_as_float(__builtin_amdgcn_update_dpp(0, __float_as_int(x), 0x140, 0xf, 0xf, true));
  x += y;
  return x;
}

// --------------------------- fp32 -> bf16 convert ---------------------------
__global__ void cvt_bf16_kernel(const float* __restrict__ in,
                                uint16_t* __restrict__ out, int n) {
  int i = blockIdx.x * blockDim.x + threadIdx.x;   // handles 2 elements
  int j = 2 * i;
  if (j + 1 < n) {
    uint32_t p = (uint32_t)f2bf(in[j]) | ((uint32_t)f2bf(in[j + 1]) << 16);
    ((uint32_t*)out)[i] = p;
  } else if (j < n) {
    out[j] = f2bf(in[j]);
  }
}

// ------------------- fp32 W[k][n] -> bf16 Wt[n][k] transpose -----------------
__global__ __launch_bounds__(256) void cvt_transpose_kernel(
    const float* __restrict__ in, uint16_t* __restrict__ out) {
  __shared__ float tile[32][33];
  const int t = threadIdx.x;
  const int bk = (blockIdx.x & 31) * 32;   // k tile base
  const int bn = (blockIdx.x >> 5) * 32;   // n tile base
#pragma unroll
  for (int i = 0; i < 4; i++) {
    int idx = t + i * 256, r = idx >> 5, c = idx & 31;
    tile[r][c] = in[(size_t)(bk + r) * D_MODEL + bn + c];
  }
  __syncthreads();
#pragma unroll
  for (int i = 0; i < 4; i++) {
    int idx = t + i * 256, r = idx >> 5, c = idx & 31;
    out[(size_t)(bn + r) * D_MODEL + bk + c] = f2bf(tile[c][r]);
  }
}

// ------------------------------- GEMM + bias --------------------------------
// C[ROWS x 1024] = A_bf16[ROWS x 1024] * Wt_bf16[1024(n) x 1024(k)]^T + bias,
// then *oscale.  Double-buffered async K pipeline (K-step 32 = one WMMA k).
// MODE 0: scatter bf16 into [B, H, S, Dk]      (Q, K projections)
// MODE 1: fp32 row-major into d_out            (output projection)
// MODE 2: scatter bf16 into [B, H, Dk, S]      (V projection, transposed)
template <int MODE>
__global__ __launch_bounds__(256) void gemm_bias_kernel(
    const uint16_t* __restrict__ A, const uint16_t* __restrict__ Bt,
    const float* __restrict__ bias, float oscale, void* __restrict__ outp) {
  __shared__ __align__(16) uint32_t ldsA[2][128 * 20];  // [row][k2], pitch 20
  __shared__ __align__(16) uint32_t ldsB[2][128 * 20];  // [n][k2],  pitch 20

  const int t = threadIdx.x;
  const int lane = t & 31;
  const int wave = t >> 5;
  const int waveM = wave & 3;   // rows 32*waveM .. +31
  const int waveN = wave >> 2;  // cols 64*waveN .. +63
  const int bm = blockIdx.x % (ROWS / 128);
  const int bn = blockIdx.x / (ROWS / 128);
  const int rowBase = bm * 128;
  const int colBase = bn * 128;
  const int mlane = lane & 15, kh = lane >> 4;

  const uint32_t* A32 = (const uint32_t*)A;   // row stride 512 u32
  const uint32_t* B32 = (const uint32_t*)Bt;  // row stride 512 u32 (n-major)

  // Copy one 128x32 bf16 tile pair into LDS buffer `buf`.
  auto copyTiles = [&](int kb, int buf) {
#pragma unroll
    for (int i = 0; i < 2; i++) {
      int idx = t + i * 256, r = idx >> 2, c4 = idx & 3;
      const uint32_t* ga = &A32[(size_t)(rowBase + r) * 512 + (kb >> 1) + c4 * 4];
      const uint32_t* gb = &B32[(size_t)(colBase + r) * 512 + (kb >> 1) + c4 * 4];
      uint32_t* la = &ldsA[buf][r * 20 + c4 * 4];
      uint32_t* lb = &ldsB[buf][r * 20 + c4 * 4];
#if HAVE_ASYNC_LDS
      async_copy16(ga, la);
      async_copy16(gb, lb);
#else
      *(uint4*)la = *(const uint4*)ga;
      *(uint4*)lb = *(const uint4*)gb;
#endif
    }
  };

  FragF acc[2][4];
#pragma unroll
  for (int i = 0; i < 2; i++)
#pragma unroll
    for (int j = 0; j < 4; j++)
#pragma unroll
      for (int e = 0; e < 8; e++) acc[i][j].f[e] = 0.0f;

  copyTiles(0, 0);
#if HAVE_ASYNC_LDS
  wait_async0();
#endif
  __syncthreads();

  for (int kb = 0; kb < D_MODEL; kb += 32) {
    const int cur = (kb >> 5) & 1;
    const bool more = (kb + 32) < D_MODEL;
#if HAVE_ASYNC_LDS
    if (more) copyTiles(kb + 32, cur ^ 1);  // async, no data registers
#endif

    FragBF afr[2];
#pragma unroll
    for (int ti = 0; ti < 2; ti++) {
      int base = (waveM * 32 + ti * 16 + mlane) * 20;
      afr[ti].q[0] = *(const uint4*)&ldsA[cur][base + kh * 4];
      afr[ti].q[1] = *(const uint4*)&ldsA[cur][base + 8 + kh * 4];
    }
#pragma unroll
    for (int tj = 0; tj < 4; tj++) {
      FragBF bfr;
      int base = (waveN * 64 + tj * 16 + mlane) * 20;
      bfr.q[0] = *(const uint4*)&ldsB[cur][base + kh * 4];
      bfr.q[1] = *(const uint4*)&ldsB[cur][base + 8 + kh * 4];
#pragma unroll
      for (int ti = 0; ti < 2; ti++) {
        acc[ti][tj].v = __builtin_amdgcn_wmma_f32_16x16x32_bf16(
            false, afr[ti].v, false, bfr.v, (short)0, acc[ti][tj].v, false,
            false);
      }
    }
    if (more) {
#if HAVE_ASYNC_LDS
      wait_async0();
#else
      copyTiles(kb + 32, cur ^ 1);          // transient-register copy
#endif
      __syncthreads();
    }
  }

  // Epilogue: C layout -> lane col n=lane&15, VGPR r -> row r+8*(lane>>4)
#pragma unroll
  for (int ti = 0; ti < 2; ti++) {
#pragma unroll
    for (int tj = 0; tj < 4; tj++) {
      int col = colBase + waveN * 64 + tj * 16 + mlane;
      float bv = bias[col];
#pragma unroll
      for (int r = 0; r < 8; r++) {
        int row = rowBase + waveM * 32 + ti * 16 + r + 8 * kh;
        float val = (acc[ti][tj].f[r] + bv) * oscale;
        int b = row >> 11, s = row & (SEQ - 1);
        int h = col >> 6, d = col & (D_K - 1);
        if (MODE == 0) {
          ((uint16_t*)outp)[((size_t)((b * NUM_HEADS + h) * SEQ + s)) * D_K +
                            d] = f2bf(val);
        } else if (MODE == 2) {
          ((uint16_t*)outp)[((size_t)((b * NUM_HEADS + h) * D_K + d)) * SEQ +
                            s] = f2bf(val);
        } else {
          ((float*)outp)[(size_t)row * D_MODEL + col] = val;
        }
      }
    }
  }
}

// --------------------------- flash attention core ---------------------------
// One block = 4 waves = one 64-row Q tile of one (b,h). Streams 64-key blocks
// with double-buffered async K/V staging. Q,K in [B,H,S,Dk]; V in [B,H,Dk,S].
__global__ __launch_bounds__(128) void attn_kernel(
    const uint16_t* __restrict__ Q, const uint16_t* __restrict__ K,
    const uint16_t* __restrict__ Vt, uint16_t* __restrict__ O) {
  __shared__ __align__(16) uint32_t ldsQ[64 * 36];        // [m][d2], pitch 36
  __shared__ __align__(16) uint32_t ldsK[2][64 * 36];     // [key][d2]
  __shared__ __align__(16) uint32_t ldsV[2][64 * 36];     // [d][key2]
  __shared__ __align__(16) uint32_t ldsP[4][16 * 36];     // per-wave P rows

  const int t = threadIdx.x, lane = t & 31, wave = t >> 5;
  const int nq = SEQ / 64;
  const int qt = blockIdx.x % nq;
  const int bh = blockIdx.x / nq;          // b*NUM_HEADS + h
  const int qbase = qt * 64;
  const int mlane = lane & 15, kh = lane >> 4;

  const uint32_t* Q32 = (const uint32_t*)Q + (size_t)bh * SEQ * 32;
  const uint32_t* K32 = (const uint32_t*)K + (size_t)bh * SEQ * 32;
  const uint32_t* V32 = (const uint32_t*)Vt + (size_t)bh * D_K * (SEQ / 2);

  auto copyKV = [&](int kv, int buf) {
#pragma unroll
    for (int i = 0; i < 4; i++) {
      int idx = t + i * 128, r = idx >> 3, c4 = idx & 7;
      const uint32_t* gk = &K32[(size_t)(kv + r) * 32 + c4 * 4];
      const uint32_t* gv = &V32[(size_t)r * (SEQ / 2) + (kv >> 1) + c4 * 4];
      uint32_t* lk = &ldsK[buf][r * 36 + c4 * 4];
      uint32_t* lv = &ldsV[buf][r * 36 + c4 * 4];
#if HAVE_ASYNC_LDS
      async_copy16(gk, lk);
      async_copy16(gv, lv);
#else
      *(uint4*)lk = *(const uint4*)gk;
      *(uint4*)lv = *(const uint4*)gv;
#endif
    }
  };

  // Stage Q tile + first K/V block, one barrier
#pragma unroll
  for (int i = 0; i < 4; i++) {
    int idx = t + i * 128, r = idx >> 3, c4 = idx & 7;
    *(uint4*)&ldsQ[r * 36 + c4 * 4] =
        *(const uint4*)&Q32[(size_t)(qbase + r) * 32 + c4 * 4];
  }
  copyKV(0, 0);
#if HAVE_ASYNC_LDS
  wait_async0();
#endif
  __syncthreads();

  FragBF qf[2];
#pragma unroll
  for (int ks = 0; ks < 2; ks++) {
    int base = (wave * 16 + mlane) * 36 + ks * 16;
    qf[ks].q[0] = *(const uint4*)&ldsQ[base + kh * 4];
    qf[ks].q[1] = *(const uint4*)&ldsQ[base + 8 + kh * 4];
  }

  float m_i[8], l_i[8];
  FragF o_acc[4];
#pragma unroll
  for (int r = 0; r < 8; r++) { m_i[r] = -1e30f; l_i[r] = 0.0f; }
#pragma unroll
  for (int j = 0; j < 4; j++)
#pragma unroll
    for (int e = 0; e < 8; e++) o_acc[j].f[e] = 0.0f;

  for (int kv = 0; kv < SEQ; kv += 64) {
    const int cur = (kv >> 6) & 1;
    const bool more = (kv + 64) < SEQ;
#if HAVE_ASYNC_LDS
    if (more) copyKV(kv + 64, cur ^ 1);    // async prefetch, no registers
#endif

    // S = Q * K^T : 4 key tiles x 2 k-steps of 32
    FragF sfr[4];
#pragma unroll
    for (int tj = 0; tj < 4; tj++)
#pragma unroll
      for (int e = 0; e < 8; e++) sfr[tj].f[e] = 0.0f;
#pragma unroll
    for (int ks = 0; ks < 2; ks++)
#pragma unroll
      for (int tj = 0; tj < 4; tj++) {
        FragBF bfr;
        int base = (tj * 16 + mlane) * 36 + ks * 16;
        bfr.q[0] = *(const uint4*)&ldsK[cur][base + kh * 4];
        bfr.q[1] = *(const uint4*)&ldsK[cur][base + 8 + kh * 4];
        sfr[tj].v = __builtin_amdgcn_wmma_f32_16x16x32_bf16(
            false, qf[ks].v, false, bfr.v, (short)0, sfr[tj].v, false, false);
      }

    // Online softmax (scale folded into Q); DPP16 row reductions.
    float mnew[8], alpha[8];
#pragma unroll
    for (int r = 0; r < 8; r++) {
      float mx = fmaxf(fmaxf(sfr[0].f[r], sfr[1].f[r]),
                       fmaxf(sfr[2].f[r], sfr[3].f[r]));
      mx = red16_max(mx);
      mnew[r] = fmaxf(m_i[r], mx);
      alpha[r] = __expf(m_i[r] - mnew[r]);
    }
#pragma unroll
    for (int r = 0; r < 8; r++) {
      float rs = 0.0f;
#pragma unroll
      for (int tj = 0; tj < 4; tj++) {
        float p = __expf(sfr[tj].f[r] - mnew[r]);
        sfr[tj].f[r] = p;
        rs += p;
      }
      rs = red16_sum(rs);
      l_i[r] = l_i[r] * alpha[r] + rs;
      m_i[r] = mnew[r];
#pragma unroll
      for (int tn = 0; tn < 4; tn++) o_acc[tn].f[r] *= alpha[r];
    }

    // Stage P bf16 (per-wave region; wave-local DS ordering is enough)
#pragma unroll
    for (int r = 0; r < 8; r++) {
      int rl = r + 8 * kh;
#pragma unroll
      for (int tj = 0; tj < 4; tj++)
        ((uint16_t*)ldsP[wave])[rl * 72 + tj * 16 + mlane] = f2bf(sfr[tj].f[r]);
    }
    asm volatile("s_wait_dscnt 0x0" ::: "memory");

    // O += P * V : 2 k-steps of 32 keys, 4 d tiles
    FragBF pf[2];
#pragma unroll
    for (int ks = 0; ks < 2; ks++) {
      int base = mlane * 36 + ks * 16;
      pf[ks].q[0] = *(const uint4*)&ldsP[wave][base + kh * 4];
      pf[ks].q[1] = *(const uint4*)&ldsP[wave][base + 8 + kh * 4];
    }
#pragma unroll
    for (int ks = 0; ks < 2; ks++)
#pragma unroll
      for (int tn = 0; tn < 4; tn++) {
        FragBF bfr;
        int base = (tn * 16 + mlane) * 36 + ks * 16;
        bfr.q[0] = *(const uint4*)&ldsV[cur][base + kh * 4];
        bfr.q[1] = *(const uint4*)&ldsV[cur][base + 8 + kh * 4];
        o_acc[tn].v = __builtin_amdgcn_wmma_f32_16x16x32_bf16(
            false, pf[ks].v, false, bfr.v, (short)0, o_acc[tn].v, false, false);
      }

    if (more) {
#if HAVE_ASYNC_LDS
      wait_async0();
#else
      copyKV(kv + 64, cur ^ 1);            // transient-register copy
#endif
      __syncthreads();
    }
  }

  // Finalize: O /= l, write bf16 into [b, s, h*64+d] for the final projection
  const int b = bh / NUM_HEADS, h = bh % NUM_HEADS;
#pragma unroll
  for (int r = 0; r < 8; r++) {
    float inv = 1.0f / l_i[r];
    int s = qbase + wave * 16 + r + 8 * kh;
#pragma unroll
    for (int tn = 0; tn < 4; tn++) {
      int d = tn * 16 + mlane;
      O[((size_t)(b * SEQ + s)) * D_MODEL + h * D_K + d] =
          f2bf(o_acc[tn].f[r] * inv);
    }
  }
}

// -------------------------------- launcher ----------------------------------
extern "C" void kernel_launch(void* const* d_in, const int* in_sizes, int n_in,
                              void* d_out, int out_size, void* d_ws,
                              size_t ws_size, hipStream_t stream) {
  (void)in_sizes; (void)n_in; (void)out_size; (void)ws_size;
  const float* x  = (const float*)d_in[0];
  const float* Wq = (const float*)d_in[1];
  const float* bq = (const float*)d_in[2];
  const float* Wk = (const float*)d_in[3];
  const float* bk = (const float*)d_in[4];
  const float* Wv = (const float*)d_in[5];
  const float* bv = (const float*)d_in[6];
  const float* Wo = (const float*)d_in[7];
  const float* bo = (const float*)d_in[8];

  char* ws = (char*)d_ws;
  const size_t SZX = (size_t)ROWS * D_MODEL * 2;      // 8 MB bf16
  const size_t SZW = (size_t)D_MODEL * D_MODEL * 2;   // 2 MB bf16
  uint16_t* xb  = (uint16_t*)(ws);
  uint16_t* wqt = (uint16_t*)(ws + SZX);
  uint16_t* wkt = (uint16_t*)(ws + SZX + SZW);
  uint16_t* wvt = (uint16_t*)(ws + SZX + 2 * SZW);
  uint16_t* wot = (uint16_t*)(ws + SZX + 3 * SZW);
  uint16_t* qb  = (uint16_t*)(ws + SZX + 4 * SZW);      // [B,H,S,Dk]
  uint16_t* kb  = (uint16_t*)(ws + 2 * SZX + 4 * SZW);  // [B,H,S,Dk]
  uint16_t* vb  = (uint16_t*)(ws + 3 * SZX + 4 * SZW);  // [B,H,Dk,S]
  uint16_t* ab  = (uint16_t*)(ws + 4 * SZX + 4 * SZW);  // [B,S,D_MODEL]

  const int nX = ROWS * D_MODEL;
  cvt_bf16_kernel<<<(nX / 2 + 255) / 256, 256, 0, stream>>>(x, xb, nX);
  dim3 tgrid(32 * 32);
  cvt_transpose_kernel<<<tgrid, 256, 0, stream>>>(Wq, wqt);
  cvt_transpose_kernel<<<tgrid, 256, 0, stream>>>(Wk, wkt);
  cvt_transpose_kernel<<<tgrid, 256, 0, stream>>>(Wv, wvt);
  cvt_transpose_kernel<<<tgrid, 256, 0, stream>>>(Wo, wot);

  dim3 ggrid((ROWS / 128) * (D_MODEL / 128));  // 256 blocks
  // fold 1/sqrt(Dk)=0.125 into Q so softmax needs no per-score multiply
  gemm_bias_kernel<0><<<ggrid, 256, 0, stream>>>(xb, wqt, bq, 0.125f, qb);
  gemm_bias_kernel<0><<<ggrid, 256, 0, stream>>>(xb, wkt, bk, 1.0f, kb);
  gemm_bias_kernel<2><<<ggrid, 256, 0, stream>>>(xb, wvt, bv, 1.0f, vb);

  attn_kernel<<<BATCH * NUM_HEADS * (SEQ / 64), 128, 0, stream>>>(qb, kb, vb,
                                                                  ab);

  gemm_bias_kernel<1><<<ggrid, 256, 0, stream>>>(ab, wot, bo, 1.0f, d_out);
}
